// PointNet_Estimation_67242007986177
// MI455X (gfx1250) — compile-verified, matrix operand order checked by reading, not run on hardware
//
#include <hip/hip_runtime.h>
#include <hip/hip_bf16.h>

typedef __attribute__((ext_vector_type(16))) _Float16 v16h;
typedef __attribute__((ext_vector_type(8)))  float    v8f;
typedef __attribute__((ext_vector_type(4)))  int      v4i;
typedef _Float16 half_t;

__host__ __device__ constexpr int pad32(int x) { return (x + 31) & ~31; }

// CDNA5 async global->LDS path (builtin confirmed present by round-2 diagnostic:
// signature takes addrspace(1)/(3) pointers to 16B int vectors).
#if defined(__has_builtin)
#if __has_builtin(__builtin_amdgcn_global_load_async_to_lds_b128) && \
    __has_builtin(__builtin_amdgcn_s_wait_asynccnt)
#define HAVE_ASYNC_LDS 1
#endif
#endif

// ---------------- WMMA fragment helpers (wave32, 16x16x32 f16) ----------------
// A-matrix 16x32 layout (ISA 7.12.2): lanes 0-15 row M=lane hold K=0..7,16..23;
// lanes 16-31 row M=lane-16 hold K=8..15,24..31.  B mirrored with N in place of
// M, so a row-major (cout x ldk) W' loads exactly like row-major A.
__device__ __forceinline__ v16h load_frag(const half_t* base, int ld, int row0, int k0) {
  const int lane = threadIdx.x & 31;
  const int r  = row0 + (lane & 15);
  const int kb = k0 + ((lane >> 4) << 3);
  const half_t* p = base + (long)r * ld + kb;
  v16h v;
#pragma unroll
  for (int j = 0; j < 8; ++j) { v[j] = p[j]; v[8 + j] = p[16 + j]; }
  return v;
}

__device__ __forceinline__ v16h load_frag_clamp(const half_t* base, int ld, int row0,
                                                int maxrow, int k0) {
  const int lane = threadIdx.x & 31;
  int r = row0 + (lane & 15);
  if (r >= maxrow) r = 0;   // safe row; result discarded by store guard
  const int kb = k0 + ((lane >> 4) << 3);
  const half_t* p = base + (long)r * ld + kb;
  v16h v;
#pragma unroll
  for (int j = 0; j < 8; ++j) { v[j] = p[j]; v[8 + j] = p[16 + j]; }
  return v;
}

__device__ __forceinline__ v8f wmma_step(v16h a, v16h b, v8f c) {
  return __builtin_amdgcn_wmma_f32_16x16x32_f16(false, a, false, b, (short)0, c, false, false);
}

// ---------------- BN-fold: W' = W * gamma*rsqrt(var+eps) (f16, K padded), b' ----------------
__global__ void k_fold(const float* __restrict__ W, const float* __restrict__ bb,
                       const float* __restrict__ gamma, const float* __restrict__ beta,
                       const float* __restrict__ mean, const float* __restrict__ var,
                       int cout, int cin, int ldk,
                       half_t* __restrict__ Wh, float* __restrict__ bh) {
  long t = (long)blockIdx.x * blockDim.x + threadIdx.x;
  if (t >= (long)cout * ldk) return;
  const int o = (int)(t / ldk), k = (int)(t % ldk);
  const float s = gamma[o] * rsqrtf(var[o] + 1e-5f);
  Wh[t] = (half_t)((k < cin) ? W[(long)o * cin + k] * s : 0.f);
  if (k == 0) bh[o] = (bb[o] - mean[o]) * s + beta[o];
}

// ---------------- small data-movement kernels ----------------
__global__ void k_transpose_xyz(const float* __restrict__ in, float* __restrict__ out,
                                int N, int total) {
  int t = blockIdx.x * blockDim.x + threadIdx.x;
  if (t >= total) return;
  const int c = t % 3; const int r = t / 3; const int n = r % N; const int b = r / N;
  out[t] = in[((long)b * 3 + c) * N + n];
}

__global__ void k_dup6(const float* __restrict__ xyz, float* __restrict__ out, long total) {
  long t = (long)blockIdx.x * blockDim.x + threadIdx.x;
  if (t >= total) return;
  const int c = (int)(t % 6); const long r = t / 6;
  out[t] = xyz[r * 3 + (c % 3)];
}

__global__ void k_gather3(const float* __restrict__ xyz, int N, const int* __restrict__ idx,
                          int S, int total, float* __restrict__ out) {
  int t = blockIdx.x * blockDim.x + threadIdx.x;
  if (t >= total) return;
  const int c = t % 3; const int r = t / 3; const int b = r / S;
  out[t] = xyz[((long)b * N + idx[r]) * 3 + c];
}

// ---------------- farthest point sampling (one workgroup per batch) ----------------
__global__ void __launch_bounds__(256)
k_fps(const float* __restrict__ xyz, int N, int npoint, int* __restrict__ out) {
  __shared__ float dist[4096];
  __shared__ float rmax[256];
  __shared__ int   ridx[256];
  __shared__ float cen[3];
  __shared__ int   curs;
  const int b = blockIdx.x, tid = threadIdx.x;
  const float* X = xyz + (long)b * N * 3;
  for (int i = tid; i < N; i += 256) dist[i] = 1e10f;
  if (tid == 0) curs = 0;
  __syncthreads();
  for (int it = 0; it < npoint; ++it) {
    const int far = curs;
    if (tid == 0) out[(long)b * npoint + it] = far;
    if (tid < 3) cen[tid] = X[(long)far * 3 + tid];
    __syncthreads();
    const float cx = cen[0], cy = cen[1], cz = cen[2];
    float bm = -1.f; int bi = 0;
    for (int i = tid; i < N; i += 256) {
      const float dx = X[i * 3] - cx, dy = X[i * 3 + 1] - cy, dz = X[i * 3 + 2] - cz;
      const float d = dx * dx + dy * dy + dz * dz;
      float dd = dist[i];
      if (d < dd) { dd = d; dist[i] = d; }
      if (dd > bm) { bm = dd; bi = i; }
    }
    rmax[tid] = bm; ridx[tid] = bi;
    __syncthreads();
    for (int off = 128; off > 0; off >>= 1) {
      if (tid < off && rmax[tid + off] > rmax[tid]) {
        rmax[tid] = rmax[tid + off]; ridx[tid] = ridx[tid + off];
      }
      __syncthreads();
    }
    if (tid == 0) curs = ridx[0];
    __syncthreads();
  }
}

// ---------------- ball query: one wave per query, ordered append via ballot ----------------
__global__ void __launch_bounds__(256)
k_query_ball(const float* __restrict__ xyz, int N, const float* __restrict__ centers,
             int total, int S, float r2, int K, int* __restrict__ gidx) {
  const int gw = blockIdx.x * 8 + (threadIdx.x >> 5);
  const int lane = threadIdx.x & 31;
  if (gw >= total) return;
  const int b = gw / S;
  const float* X = xyz + (long)b * N * 3;
  const float cx = centers[(long)gw * 3 + 0];
  const float cy = centers[(long)gw * 3 + 1];
  const float cz = centers[(long)gw * 3 + 2];
  int* g = gidx + (long)gw * K;
  int cnt = 0, first = -1;
  for (int base = 0; base < N; base += 32) {
    if (cnt >= K) break;
    const int i = base + lane;
    const float dx = X[i * 3 + 0] - cx, dy = X[i * 3 + 1] - cy, dz = X[i * 3 + 2] - cz;
    const bool ok = (dx * dx + dy * dy + dz * dz) <= r2;
    const unsigned long long m = __ballot(ok);
    if (ok) {
      const int pos = cnt + (int)__popcll(m & ((1ull << lane) - 1ull));
      if (pos < K) g[pos] = i;
    }
    if (first < 0 && m != 0ull) first = base + (__ffsll((long long)m) - 1);
    cnt += (int)__popcll(m);
  }
  const int cw = cnt < K ? cnt : K;
  for (int j = cw + lane; j < K; j += 32) g[j] = first;
}

// ---------------- one MLP layer inside LDS (WMMA, 128 threads = 4 waves) ----------------
template <int RCH>
__device__ __forceinline__ void mlp_layer(const half_t* X, int ldx, half_t* Y, int ldy,
                                          const half_t* __restrict__ W,
                                          const float* __restrict__ bias, int cout) {
  const int tid = threadIdx.x;
  for (int e = tid; e < RCH * ldy; e += 128) Y[e] = (half_t)0.f;
  __syncthreads();
  const int ksteps = ldx >> 5;
  const int ntiles = (cout + 15) >> 4;
  const int wave = tid >> 5, lane = tid & 31;
  for (int t = wave; t < (RCH >> 4) * ntiles; t += 4) {
    const int rt = t / ntiles, nt = t % ntiles;
    v8f acc = {};
    for (int k = 0; k < ksteps; ++k) {
      v16h a  = load_frag(X, ldx, rt * 16, k * 32);
      v16h bm = load_frag_clamp(W, ldx, nt * 16, cout, k * 32);
      acc = wmma_step(a, bm, acc);
    }
    const int n  = nt * 16 + (lane & 15);
    const int m0 = rt * 16 + ((lane >> 4) << 3);
    if (n < cout) {
      const float bv = bias[n];
#pragma unroll
      for (int r = 0; r < 8; ++r) {
        float v = acc[r] + bv;
        Y[(m0 + r) * ldy + n] = (half_t)(v > 0.f ? v : 0.f);
      }
    }
  }
  __syncthreads();
}

// ---------------- fused gather + 3-layer MLP + maxpool, all in LDS ----------------
template <int RCH, int CAPA, int CAPB>
__global__ void __launch_bounds__(128)
k_group_mlp(const float* __restrict__ xyzsrc, const float* __restrict__ featsrc,
            int Cf, int Nsrc, const float* __restrict__ centers,
            const int* __restrict__ gidx, int S, int K,
            const half_t* __restrict__ W0, const float* __restrict__ b0, int c0,
            const half_t* __restrict__ W1, const float* __restrict__ b1, int c1,
            const half_t* __restrict__ W2, const float* __restrict__ b2, int c2,
            float* __restrict__ outfeat, int ldout, int outoff) {
  __shared__ half_t bufA[RCH * CAPA];
  __shared__ half_t bufB[RCH * CAPB];
  __shared__ float  bmax[CAPB];
  const int tid = threadIdx.x;
  const int s = blockIdx.x % S, b = blockIdx.x / S;
  const int cin = Cf + 3;
  const int ld0 = pad32(cin), ld1 = pad32(c0), ld2 = pad32(c1), ld3 = pad32(c2);
  for (int c = tid; c < CAPB; c += 128) bmax[c] = -3.4e38f;
  const float cx = centers[((long)b * S + s) * 3 + 0];
  const float cy = centers[((long)b * S + s) * 3 + 1];
  const float cz = centers[((long)b * S + s) * 3 + 2];
  const int* gi = gidx + ((long)b * S + s) * K;
  const int nch = (K + RCH - 1) / RCH;
  for (int ch = 0; ch < nch; ++ch) {
    const int base = ch * RCH;
    const int valid = (K - base) < RCH ? (K - base) : RCH;
    __syncthreads();
    for (int e = tid; e < RCH * ld0; e += 128) {
      const int r = e / ld0, c = e % ld0;
      float v = 0.f;
      if (r < valid && c < cin) {
        const int idx = gi[base + r];
        if (c < Cf) {
          v = featsrc[((long)b * Nsrc + idx) * Cf + c];
        } else {
          const int d = c - Cf;
          v = xyzsrc[((long)b * Nsrc + idx) * 3 + d] - (d == 0 ? cx : (d == 1 ? cy : cz));
        }
      }
      bufA[e] = (half_t)v;
    }
    __syncthreads();
    mlp_layer<RCH>(bufA, ld0, bufB, ld1, W0, b0, c0);
    mlp_layer<RCH>(bufB, ld1, bufA, ld2, W1, b1, c1);
    mlp_layer<RCH>(bufA, ld2, bufB, ld3, W2, b2, c2);
    for (int c = tid; c < c2; c += 128) {
      float m = bmax[c];
      for (int r = 0; r < valid; ++r) {
        const float v = (float)bufB[r * ld3 + c];
        m = v > m ? v : m;
      }
      bmax[c] = m;
    }
  }
  __syncthreads();
  for (int c = tid; c < c2; c += 128)
    outfeat[((long)b * S + s) * ldout + outoff + c] = bmax[c];
}

// ---------------- generic WMMA GEMM: O = relu(A @ W'^T + b'), f16 in/out ----------------
// Block = 64 rows x 128 cols, 8 waves. Wave w owns column tile n0 = bn + 16*w and
// all 4 row tiles (4 independent accumulators; W fragment loaded once per k-step).
// A panel staged through LDS in 32-k chunks, double buffered via async
// global->LDS (ASYNCcnt), with a plain-copy fallback.
// NOTE: all couts in this model are multiples of 128 -> no column clamp needed.
__global__ void __launch_bounds__(256)
k_gemm_bias_relu(const half_t* __restrict__ A, int lda, const half_t* __restrict__ W,
                 const float* __restrict__ bias, int cout, half_t* __restrict__ O,
                 int ldo, float* __restrict__ O32) {
  __shared__ half_t Abuf[2][64 * 32];
  const int bm = blockIdx.x * 64;
  const int bn = blockIdx.y * 128;
  const int tid = threadIdx.x, wave = tid >> 5, lane = tid & 31;
  const int ksteps = lda >> 5;
  const int n0 = bn + wave * 16;
  // Staging: thread tid copies one 16B piece per chunk: row = tid>>2, piece = tid&3.
  const int srow = tid >> 2, spiece = tid & 3;
  const char* gsrc_row = (const char*)(A + (long)(bm + srow) * lda) + spiece * 16;

  auto stage = [&](int k, int buf) {
    const char* src = gsrc_row + (long)k * 64;
    half_t* dst = &Abuf[buf][srow * 32 + spiece * 8];
#ifdef HAVE_ASYNC_LDS
    __builtin_amdgcn_global_load_async_to_lds_b128(
        (__attribute__((address_space(1))) v4i*)src,
        (__attribute__((address_space(3))) v4i*)dst, 0, 0);
#else
    *(uint4*)dst = *(const uint4*)src;
#endif
  };

  v8f acc[4];
#pragma unroll
  for (int rt = 0; rt < 4; ++rt) acc[rt] = (v8f){};

  stage(0, 0);
  for (int k = 0; k < ksteps; ++k) {
    const int cur = k & 1;
    if (k + 1 < ksteps) stage(k + 1, cur ^ 1);
#ifdef HAVE_ASYNC_LDS
    if (k + 1 < ksteps) __builtin_amdgcn_s_wait_asynccnt(1);
    else                __builtin_amdgcn_s_wait_asynccnt(0);
#endif
    __syncthreads();                       // chunk k resident & visible
    const v16h bf = load_frag(W, lda, n0, k * 32);
    v16h a0 = load_frag(Abuf[cur], 32,  0, 0);
    v16h a1 = load_frag(Abuf[cur], 32, 16, 0);
    v16h a2 = load_frag(Abuf[cur], 32, 32, 0);
    v16h a3 = load_frag(Abuf[cur], 32, 48, 0);
    acc[0] = wmma_step(a0, bf, acc[0]);
    acc[1] = wmma_step(a1, bf, acc[1]);
    acc[2] = wmma_step(a2, bf, acc[2]);
    acc[3] = wmma_step(a3, bf, acc[3]);
    __syncthreads();                       // done reading before buffer reuse
  }

  const int n = n0 + (lane & 15);
  const float bv = bias[n];
  const int mrow = (lane >> 4) << 3;
#pragma unroll
  for (int rt = 0; rt < 4; ++rt) {
    const int m = bm + rt * 16 + mrow;
#pragma unroll
    for (int r = 0; r < 8; ++r) {
      float v = acc[rt][r] + bv;
      v = v > 0.f ? v : 0.f;
      O[(long)(m + r) * ldo + n] = (half_t)v;
      if (O32) O32[(long)(m + r) * cout + n] = v;
    }
  }
}

// ---------------- 3-NN (thread per query) ----------------
__global__ void k_knn3(const float* __restrict__ xyz1, int N1, const float* __restrict__ xyz2,
                       int N2, int* __restrict__ idx3, float* __restrict__ w3, int Btot) {
  long t = (long)blockIdx.x * blockDim.x + threadIdx.x;
  if (t >= (long)Btot * N1) return;
  const int b = (int)(t / N1);
  const float* p = xyz1 + t * 3;
  const float px = p[0], py = p[1], pz = p[2];
  const float* q = xyz2 + (long)b * N2 * 3;
  float d0 = 1e30f, d1 = 1e30f, d2 = 1e30f;
  int i0 = 0, i1 = 0, i2 = 0;
  for (int s = 0; s < N2; ++s) {
    const float dx = q[s * 3] - px, dy = q[s * 3 + 1] - py, dz = q[s * 3 + 2] - pz;
    const float d = dx * dx + dy * dy + dz * dz;
    if (d < d0)      { d2 = d1; i2 = i1; d1 = d0; i1 = i0; d0 = d; i0 = s; }
    else if (d < d1) { d2 = d1; i2 = i1; d1 = d; i1 = s; }
    else if (d < d2) { d2 = d; i2 = s; }
  }
  const float w0 = 1.f / (d0 + 1e-8f), w1 = 1.f / (d1 + 1e-8f), w2 = 1.f / (d2 + 1e-8f);
  const float ws = w0 + w1 + w2;
  idx3[t * 3] = i0; idx3[t * 3 + 1] = i1; idx3[t * 3 + 2] = i2;
  w3[t * 3] = w0 / ws; w3[t * 3 + 1] = w1 / ws; w3[t * 3 + 2] = w2 / ws;
}

// ---------------- build f16 activation X = [src1 | interp(src2)] (zero K-pad) ----------------
__global__ void k_build_x(half_t* __restrict__ X, int ldk, int N,
                          const float* __restrict__ src1, int C1,
                          const float* __restrict__ src2, int C2, int S2,
                          const int* __restrict__ idx3, const float* __restrict__ w3,
                          long total) {
  long t = (long)blockIdx.x * blockDim.x + threadIdx.x;
  if (t >= total) return;
  const int c = (int)(t % ldk);
  const long rn = t / ldk;
  const int n = (int)(rn % N);
  const int b = (int)(rn / N);
  float v = 0.f;
  if (c < C1) {
    v = src1[rn * C1 + c];
  } else if (c < C1 + C2) {
    const int c2 = c - C1;
    if (idx3) {
      const int* ii = idx3 + rn * 3;
      const float* ww = w3 + rn * 3;
      const float* s2 = src2 + (long)b * S2 * C2;
      v = ww[0] * s2[(long)ii[0] * C2 + c2] + ww[1] * s2[(long)ii[1] * C2 + c2] +
          ww[2] * s2[(long)ii[2] * C2 + c2];
    } else {
      const int row = (S2 == 1) ? 0 : n;
      v = src2[((long)b * S2 + row) * C2 + c2];
    }
  }
  X[t] = (half_t)v;
}

__global__ void k_maxpool_rows(const half_t* __restrict__ act, int ld, int rows, int C,
                               int total, float* __restrict__ out) {
  int t = blockIdx.x * blockDim.x + threadIdx.x;
  if (t >= total) return;
  const int c = t % C; const int b = t / C;
  float m = -3.4e38f;
  for (int r = 0; r < rows; ++r) {
    const float v = (float)act[((long)b * rows + r) * ld + c];
    m = v > m ? v : m;
  }
  out[t] = m;
}

__global__ void k_out(const half_t* __restrict__ act, float* __restrict__ out, int N, int C,
                      long total) {
  long t = (long)blockIdx.x * blockDim.x + threadIdx.x;
  if (t >= total) return;
  const int n = (int)(t % N);
  const long r = t / N;
  const int c = (int)(r % C);
  const int b = (int)(r / C);
  out[t] = (float)act[((long)b * N + n) * C + c];
}

// ======================= host orchestration =======================
extern "C" void kernel_launch(void* const* d_in, const int* in_sizes, int n_in,
                              void* d_out, int out_size, void* d_ws, size_t ws_size,
                              hipStream_t stream) {
  (void)in_sizes; (void)n_in; (void)out_size; (void)ws_size;
  const int B = 8, N = 4096;
  // Layer tables (traversal order: sa1 b0..b2, sa2 b0..b1, sa3, fp3, fp2, fp1);
  // each layer occupies 6 consecutive inputs starting at index 2: W,b,gamma,beta,mean,var.
  static const int LCIN[24]  = {6, 32, 32,  6, 64, 64,  6, 64, 96,
                                323, 128, 128,  323, 128, 196,
                                515, 256, 512,  1536, 256,  576, 256,  134, 128};
  static const int LCOUT[24] = {32, 32, 64,  64, 64, 128,  64, 96, 128,
                                128, 128, 256,  128, 196, 256,
                                256, 512, 1024,  256, 256,  256, 128,  128, 512};

  // ---- workspace carve ----
  char* w = (char*)d_ws;
  auto alloc = [&](size_t bytes) -> void* {
    void* p = (void*)w;
    w += (bytes + 255) & ~((size_t)255);
    return p;
  };
  float* l0xyz = (float*)alloc((size_t)B * N * 3 * 4);
  float* l0cat = (float*)alloc((size_t)B * N * 6 * 4);
  half_t* Wh[24]; float* Bh[24];
  for (int t = 0; t < 24; ++t) {
    Wh[t] = (half_t*)alloc((size_t)LCOUT[t] * pad32(LCIN[t]) * 2);
    Bh[t] = (float*)alloc((size_t)LCOUT[t] * 4);
  }
  int*   fps1  = (int*)alloc((size_t)B * 512 * 4);
  float* nx1   = (float*)alloc((size_t)B * 512 * 3 * 4);
  int*   gbuf  = (int*)alloc((size_t)B * 512 * 128 * 4);
  float* feat1 = (float*)alloc((size_t)B * 512 * 320 * 4);
  int*   fps2  = (int*)alloc((size_t)B * 128 * 4);
  float* nx2   = (float*)alloc((size_t)B * 128 * 3 * 4);
  float* feat2 = (float*)alloc((size_t)B * 128 * 512 * 4);
  half_t* bufX = (half_t*)alloc((size_t)B * N * 160 * 2);
  half_t* bufP = (half_t*)alloc((size_t)B * N * 512 * 2);
  half_t* bufQ = (half_t*)alloc((size_t)B * N * 512 * 2);
  float* l3p   = (float*)alloc((size_t)B * 1024 * 4);
  float* fp3o  = (float*)alloc((size_t)B * 128 * 256 * 4);
  float* fp2o  = (float*)alloc((size_t)B * 512 * 128 * 4);
  int*   knn_i = (int*)alloc((size_t)B * N * 3 * 4);
  float* knn_w = (float*)alloc((size_t)B * N * 3 * 4);

  // ---- prep ----
  const float* xyz_in = (const float*)d_in[0];
  k_transpose_xyz<<<(B * N * 3 + 255) / 256, 256, 0, stream>>>(xyz_in, l0xyz, N, B * N * 3);
  k_dup6<<<(int)(((long)B * N * 6 + 255) / 256), 256, 0, stream>>>(l0xyz, l0cat, (long)B * N * 6);
  for (int t = 0; t < 24; ++t) {
    const float* Wp = (const float*)d_in[2 + 6 * t + 0];
    const float* bp = (const float*)d_in[2 + 6 * t + 1];
    const float* gp = (const float*)d_in[2 + 6 * t + 2];
    const float* tp = (const float*)d_in[2 + 6 * t + 3];
    const float* mp = (const float*)d_in[2 + 6 * t + 4];
    const float* vp = (const float*)d_in[2 + 6 * t + 5];
    const int ldk = pad32(LCIN[t]);
    const long tot = (long)LCOUT[t] * ldk;
    k_fold<<<(int)((tot + 255) / 256), 256, 0, stream>>>(Wp, bp, gp, tp, mp, vp, LCOUT[t],
                                                         LCIN[t], ldk, Wh[t], Bh[t]);
  }

  // ---- sa1 ----
  k_fps<<<B, 256, 0, stream>>>(l0xyz, N, 512, fps1);
  k_gather3<<<(B * 512 * 3 + 255) / 256, 256, 0, stream>>>(l0xyz, N, fps1, 512, B * 512 * 3, nx1);
  {
    const float R1[3] = {0.1f, 0.2f, 0.4f};
    const int K1[3] = {32, 64, 128}, T1[3] = {0, 3, 6}, O1[3] = {0, 64, 192};
    for (int r = 0; r < 3; ++r) {
      k_query_ball<<<(B * 512 + 7) / 8, 256, 0, stream>>>(l0xyz, N, nx1, B * 512, 512,
                                                          R1[r] * R1[r], K1[r], gbuf);
      const int t = T1[r];
      k_group_mlp<64, 96, 128><<<B * 512, 128, 0, stream>>>(
          l0xyz, l0xyz, 3, N, nx1, gbuf, 512, K1[r],
          Wh[t], Bh[t], LCOUT[t], Wh[t + 1], Bh[t + 1], LCOUT[t + 1],
          Wh[t + 2], Bh[t + 2], LCOUT[t + 2], feat1, 320, O1[r]);
    }
  }

  // ---- sa2 ----
  k_fps<<<B, 256, 0, stream>>>(nx1, 512, 128, fps2);
  k_gather3<<<(B * 128 * 3 + 255) / 256, 256, 0, stream>>>(nx1, 512, fps2, 128, B * 128 * 3, nx2);
  {
    const float R2[2] = {0.4f, 0.8f};
    const int K2[2] = {64, 128}, T2[2] = {9, 12}, O2[2] = {0, 256};
    for (int r = 0; r < 2; ++r) {
      k_query_ball<<<(B * 128 + 7) / 8, 256, 0, stream>>>(nx1, 512, nx2, B * 128, 128,
                                                          R2[r] * R2[r], K2[r], gbuf);
      const int t = T2[r];
      k_group_mlp<64, 352, 256><<<B * 128, 128, 0, stream>>>(
          nx1, feat1, 320, 512, nx2, gbuf, 128, K2[r],
          Wh[t], Bh[t], LCOUT[t], Wh[t + 1], Bh[t + 1], LCOUT[t + 1],
          Wh[t + 2], Bh[t + 2], LCOUT[t + 2], feat2, 512, O2[r]);
    }
  }

  auto gemm = [&](const half_t* A, int lda, int M, int t, half_t* O, float* O32) {
    dim3 g(M / 64, (LCOUT[t] + 127) / 128);
    k_gemm_bias_relu<<<g, 256, 0, stream>>>(A, lda, Wh[t], Bh[t], LCOUT[t], O,
                                            pad32(LCOUT[t]), O32);
  };

  // ---- sa3 (group-all) ----
  {
    const long tot = (long)B * 128 * 544;
    k_build_x<<<(int)((tot + 255) / 256), 256, 0, stream>>>(bufX, 544, 128, nx2, 3, feat2, 512,
                                                            128, nullptr, nullptr, tot);
  }
  gemm(bufX, 544, B * 128, 15, bufP, nullptr);
  gemm(bufP, 256, B * 128, 16, bufQ, nullptr);
  gemm(bufQ, 512, B * 128, 17, bufP, nullptr);
  k_maxpool_rows<<<(B * 1024 + 255) / 256, 256, 0, stream>>>(bufP, 1024, 128, 1024, B * 1024, l3p);

  // ---- fp3 ----
  {
    const long tot = (long)B * 128 * 1536;
    k_build_x<<<(int)((tot + 255) / 256), 256, 0, stream>>>(bufX, 1536, 128, feat2, 512, l3p,
                                                            1024, 1, nullptr, nullptr, tot);
  }
  gemm(bufX, 1536, B * 128, 18, bufP, nullptr);
  gemm(bufP, 256, B * 128, 19, bufQ, fp3o);

  // ---- fp2 ----
  k_knn3<<<(B * 512 + 255) / 256, 256, 0, stream>>>(nx1, 512, nx2, 128, knn_i, knn_w, B);
  {
    const long tot = (long)B * 512 * 576;
    k_build_x<<<(int)((tot + 255) / 256), 256, 0, stream>>>(bufX, 576, 512, feat1, 320, fp3o,
                                                            256, 128, knn_i, knn_w, tot);
  }
  gemm(bufX, 576, B * 512, 20, bufP, nullptr);
  gemm(bufP, 256, B * 512, 21, bufQ, fp2o);

  // ---- fp1 ----
  k_knn3<<<(B * N + 255) / 256, 256, 0, stream>>>(l0xyz, N, nx1, 512, knn_i, knn_w, B);
  {
    const long tot = (long)B * N * 160;
    k_build_x<<<(int)((tot + 255) / 256), 256, 0, stream>>>(bufX, 160, N, l0cat, 6, fp2o, 128,
                                                            512, knn_i, knn_w, tot);
  }
  gemm(bufX, 160, B * N, 22, bufP, nullptr);
  gemm(bufP, 128, B * N, 23, bufQ, nullptr);

  // ---- final transpose to (B, 512, N) f32 ----
  {
    const long tot = (long)B * 512 * N;
    k_out<<<(int)((tot + 255) / 256), 256, 0, stream>>>(bufQ, (float*)d_out, N, 512, tot);
  }
}
